// Eagle3Attention_51092930953928
// MI455X (gfx1250) — matrix operand order repaired and checked by reading, not compiled, not run on hardware
//
#include <hip/hip_runtime.h>
#include <hip/hip_bf16.h>
#include <hip/hip_fp16.h>

typedef __attribute__((ext_vector_type(16))) _Float16 v16h;
typedef __attribute__((ext_vector_type(8)))  float    v8f;
typedef __attribute__((ext_vector_type(4)))  unsigned v4u;
typedef __attribute__((ext_vector_type(8)))  unsigned v8u;

constexpr int kB   = 2;
constexpr int kS   = 2048;
constexpr int kH   = 4096;
constexpr int kNH  = 32;
constexpr int kNKV = 8;
constexpr int kHD  = 128;
constexpr int kM   = kB * kS;   // 4096 rows of X

// LDS byte offset of a __shared__ object: flat LDS-aperture addresses carry the
// LDS offset in their low 32 bits (ISA 10.2).
#define LDS_OFF(p) ((unsigned)(unsigned long long)(const void*)(p))

__device__ __forceinline__ void async_load_b128(unsigned lds_off, const void* gaddr) {
  asm volatile("global_load_async_to_lds_b128 %0, %1, off"
               :: "v"(lds_off), "v"(gaddr) : "memory");
}
__device__ __forceinline__ void wait_asynccnt0() {
  asm volatile("s_wait_asynccnt 0x0" ::: "memory");
}

// ---------------------------------------------------------------------------
// f32 -> f16 conversion
// ---------------------------------------------------------------------------
__global__ void cvt_f32_f16_kernel(const float* __restrict__ src,
                                   _Float16* __restrict__ dst, int n) {
  int i = blockIdx.x * blockDim.x + threadIdx.x;
  if (i < n) dst[i] = (_Float16)src[i];
}

// ---------------------------------------------------------------------------
// WMMA GEMM: C[M,N] = A[M,K] * B[K,N] (+bias)
// MODE 0: f16 output scattered to [B, nheads, S, HD] (QKV projections)
// MODE 1: f32 output, plain row-major [M,N] (final O-projection)
// 256 threads (8 waves), 128x128 tile, K-step 32, double-buffered LDS staged
// via CDNA5 async global->LDS DMA so copy of tile k+1 overlaps WMMAs of tile k.
// ---------------------------------------------------------------------------
template<int MODE>
__global__ __launch_bounds__(256)
void gemm_wmma_kernel(const _Float16* __restrict__ A,
                      const _Float16* __restrict__ Bm,
                      const float* __restrict__ bias,
                      _Float16* __restrict__ outh,
                      float* __restrict__ outf,
                      int Mdim, int Ndim, int Kdim, int nheads) {
  constexpr int TM = 128, TN = 128, KC = 32;
  __shared__ _Float16 As[2][TM * KC];   // [m][k], row stride 32 halfs
  __shared__ _Float16 Bs[2][KC * TN];   // [k][n], row stride 128 halfs

  const int tid  = threadIdx.x;
  const int lane = tid & 31;
  const int w    = tid >> 5;
  const int lidx = lane & 15;
  const int hi   = lane >> 4;
  const int wm   = w & 3;
  const int wn   = w >> 2;
  const int m0   = blockIdx.y * TM;
  const int n0   = blockIdx.x * TN;
  const int kb   = hi ? 8 : 0;

  v8f acc[2][4];
#pragma unroll
  for (int s = 0; s < 2; ++s)
#pragma unroll
    for (int t = 0; t < 4; ++t) acc[s][t] = (v8f){};

  // Async stage of one K-tile into buffer `buf` (16B per transfer per lane).
  auto stage = [&](int k0, int buf) {
#pragma unroll
    for (int i = tid; i < 512; i += 256) {        // A: 128 rows x 4 x 16B
      int row = i >> 2, seg = (i & 3) << 3;       // seg in halfs
      async_load_b128(LDS_OFF(&As[buf][row * KC + seg]),
                      &A[(size_t)(m0 + row) * Kdim + k0 + seg]);
    }
#pragma unroll
    for (int i = tid; i < 512; i += 256) {        // B: 32 rows x 16 x 16B
      int row = i >> 4, seg = (i & 15) << 3;
      async_load_b128(LDS_OFF(&Bs[buf][row * TN + seg]),
                      &Bm[(size_t)(k0 + row) * Ndim + n0 + seg]);
    }
  };

  const int nK = Kdim / KC;
  stage(0, 0);
  wait_asynccnt0();
  __syncthreads();

  for (int kt = 0; kt < nK; ++kt) {
    const int buf = kt & 1;
    if (kt + 1 < nK) stage((kt + 1) * KC, buf ^ 1);   // overlap DMA with WMMA

    // A fragments (ISA 16-bit A layout: lane = row, K = kb + {0..7,16..23})
    v16h af[2];
#pragma unroll
    for (int sub = 0; sub < 2; ++sub) {
      const _Float16* ab = &As[buf][(wm * 32 + sub * 16 + lidx) * KC];
      union { v16h v; unsigned int u[8]; } r;
#pragma unroll
      for (int j = 0; j < 8; ++j) {
        int k = kb + ((j < 4) ? (j << 1) : (16 + ((j - 4) << 1)));
        r.u[j] = *(const unsigned int*)(ab + k);
      }
      af[sub] = r.v;
    }
#pragma unroll
    for (int t = 0; t < 4; ++t) {
      v16h bf;
#pragma unroll
      for (int s = 0; s < 16; ++s)
        bf[s] = Bs[buf][(kb + s + ((s >> 3) << 3)) * TN + wn * 64 + t * 16 + lidx];
#pragma unroll
      for (int sub = 0; sub < 2; ++sub)
        acc[sub][t] = __builtin_amdgcn_wmma_f32_16x16x32_f16(
            false, af[sub], false, bf, (short)0, acc[sub][t], false, false);
    }
    wait_asynccnt0();
    __syncthreads();
  }

  // Epilogue: C layout -> element (row = r + 8*hi, col = lane&15)
#pragma unroll
  for (int sub = 0; sub < 2; ++sub)
#pragma unroll
    for (int t = 0; t < 4; ++t)
#pragma unroll
      for (int r = 0; r < 8; ++r) {
        int mm = m0 + wm * 32 + sub * 16 + r + 8 * hi;
        int nn = n0 + wn * 64 + t * 16 + lidx;
        float val = acc[sub][t][r];
        if (MODE == 0) {
          val += bias[nn];
          int b = mm / kS, s = mm - b * kS;
          int hh = nn >> 7, d = nn & 127;   // HD = 128
          outh[(((size_t)b * nheads + hh) * kS + s) * kHD + d] = (_Float16)val;
        } else {
          outf[(size_t)mm * Ndim + nn] = val;
        }
      }
}

// ---------------------------------------------------------------------------
// RoPE (Llama-style, pairs (d, d+64)) applied in-place on [B, nheads, S, HD] f16
// ---------------------------------------------------------------------------
__global__ void rope_kernel(_Float16* __restrict__ X,
                            const int* __restrict__ pos_ids,
                            int nheads, int total) {
  int idx = blockIdx.x * blockDim.x + threadIdx.x;
  if (idx >= total) return;
  int d    = idx & 63;
  int s    = (idx >> 6) & (kS - 1);
  int rest = idx >> 17;                 // 64 * 2048 = 2^17
  int h    = rest % nheads;
  int b    = rest / nheads;
  float pos  = (float)pos_ids[b * kS + s];
  float ang = pos * __expf(-(float)d * (9.210340371976184f / 64.0f));
  float c = __cosf(ang), sn = __sinf(ang);
  _Float16* p = X + (((size_t)b * nheads + h) * kS + s) * kHD;
  float x1 = (float)p[d], x2 = (float)p[d + 64];
  p[d]      = (_Float16)(x1 * c - x2 * sn);
  p[d + 64] = (_Float16)(x2 * c + x1 * sn);
}

// ---------------------------------------------------------------------------
// Flash attention (causal, GQA 32q/8kv heads, HD=128), WMMA f16 -> f32.
// Grid: B*NH*(S/64) blocks, 128 threads (4 waves). Wave owns 16 query rows.
// V block (contiguous 8KB) staged by the Tensor Data Mover (wave 0 issues a
// tensor_load_to_lds with a 2-group D#); K staged manually transposed.
// ---------------------------------------------------------------------------
__global__ __launch_bounds__(128)
void attn_wmma_kernel(const _Float16* __restrict__ Q,
                      const _Float16* __restrict__ K,
                      const _Float16* __restrict__ V,
                      _Float16* __restrict__ O) {
  __shared__ _Float16 KT[128 * 32];      // [d][key]  (transposed K block)
  __shared__ _Float16 Vs[32 * 128];      // [key][d]  (TDM destination)
  __shared__ _Float16 Ps[4 * 16 * 32];   // per-wave P scratch [16 q][32 key]

  const int tid  = threadIdx.x;
  const int w    = tid >> 5;
  const int lane = tid & 31;
  const int lidx = lane & 15;
  const int hi   = lane >> 4;
  const int kb   = hi ? 8 : 0;

  const int blk    = blockIdx.x;
  const int qchunk = blk & 31;           // S/64 = 32 chunks per (b,h)
  const int bh     = blk >> 5;
  const int b      = bh >> 5;            // NH = 32
  const int h      = bh & 31;
  const int kvh    = h / (kNH / kNKV);   // repeat_interleave mapping
  const int q0     = qchunk * 64 + w * 16;

  // Preload Q A-fragments for the 4 K-chunks of HD=128
  const _Float16* qbase = Q + (((size_t)b * kNH + h) * kS + q0) * kHD;
  v16h aQ[4];
#pragma unroll
  for (int c = 0; c < 4; ++c) {
    union { v16h v; unsigned int u[8]; } r;
#pragma unroll
    for (int j = 0; j < 8; ++j) {
      int d = c * 32 + kb + ((j < 4) ? (j << 1) : (16 + ((j - 4) << 1)));
      r.u[j] = *(const unsigned int*)(qbase + (size_t)lidx * kHD + d);
    }
    aQ[c] = r.v;
  }

  v8f o[8];
#pragma unroll
  for (int t = 0; t < 8; ++t) o[t] = (v8f){};
  float mrow[8], lrow[8];
#pragma unroll
  for (int r = 0; r < 8; ++r) { mrow[r] = -3.0e38f; lrow[r] = 0.0f; }

  const float scale = 0.08838834764831845f;   // 1/sqrt(128)
  const int nblocks = (qchunk + 1) * 2;       // 32-key blocks up to chunk end
  const _Float16* kbasep = K + (((size_t)b * kNKV + kvh) * kS) * kHD;
  const _Float16* vbasep = V + (((size_t)b * kNKV + kvh) * kS) * kHD;

  for (int bkk = 0; bkk < nblocks; ++bkk) {
    const int kstart = bkk * 32;

    // --- V block via Tensor Data Mover (one DMA per block, wave 0 only) ---
    if (w == 0) {
      unsigned long long ga = (unsigned long long)(vbasep + (size_t)kstart * kHD);
      v4u g0;
      g0.x = 1u;                                           // count = 1
      g0.y = LDS_OFF(&Vs[0]);                              // lds_addr
      g0.z = (unsigned)ga;                                 // global_addr[31:0]
      g0.w = ((unsigned)(ga >> 32) & 0x01FFFFFFu) | (2u << 30);  // addr[56:32] | type=2
      union { v8u v; unsigned u[8]; } g1;
      g1.u[0] = 3u << 16;            // wg_mask=0, data_size=8B, no flags
      g1.u[1] = 1024u << 16;         // tensor_dim0 = 1024 (low 16 @ bits 63:48)
      g1.u[2] = 1u << 16;            // tensor_dim0 hi=0, tensor_dim1 = 1
      g1.u[3] = 1024u << 16;         // tensor_dim1 hi=0, tile_dim0 = 1024
      g1.u[4] = 1u;                  // tile_dim1 = 1, tile_dim2 = 0
      g1.u[5] = 1024u;               // tensor_dim0_stride = 1024
      g1.u[6] = 0u;                  // stride0 hi, stride1 lo
      g1.u[7] = 0u;                  // stride1 hi
      v4u gz = (v4u){0u, 0u, 0u, 0u};
      asm volatile("tensor_load_to_lds %0, %1, %2, %3"
                   :: "s"(g0), "s"(g1.v), "s"(gz), "s"(gz) : "memory");
      __builtin_amdgcn_s_wait_tensorcnt(0);
    }

    // --- K block staged transposed (all 128 threads) ---
    for (int i = tid; i < 32 * 128; i += 128) {
      int key = i >> 7, d = i & 127;
      KT[d * 32 + key] = kbasep[(size_t)(kstart + key) * kHD + d];
    }
    // Prefetch next K/V blocks into cache (global_prefetch_b8)
    if (bkk + 1 < nblocks) {
      __builtin_prefetch(kbasep + (size_t)(kstart + 32) * kHD + tid * 32, 0, 0);
      __builtin_prefetch(vbasep + (size_t)(kstart + 32) * kHD + tid * 32, 0, 0);
    }
    __syncthreads();

    if (kstart <= q0 + 15) {              // block not entirely above the diagonal
      // S = Q * K^T  (two 16x16 tiles covering 32 keys)
      v8f s0 = (v8f){}, s1 = (v8f){};
#pragma unroll
      for (int c = 0; c < 4; ++c) {
        v16h b0, b1;
#pragma unroll
        for (int s = 0; s < 16; ++s) {
          int dk = c * 32 + kb + s + ((s >> 3) << 3);
          b0[s] = KT[dk * 32 + lidx];
          b1[s] = KT[dk * 32 + 16 + lidx];
        }
        s0 = __builtin_amdgcn_wmma_f32_16x16x32_f16(false, aQ[c], false, b0,
                                                    (short)0, s0, false, false);
        s1 = __builtin_amdgcn_wmma_f32_16x16x32_f16(false, aQ[c], false, b1,
                                                    (short)0, s1, false, false);
      }

      // Online softmax (row stats live in the same lane/vgpr slots as C rows)
      float p0a[8], p1a[8];
#pragma unroll
      for (int r = 0; r < 8; ++r) {
        int row = q0 + r + 8 * hi;
        float v0 = s0[r] * scale, v1 = s1[r] * scale;
        if (kstart + lidx > row)      v0 = -3.0e38f;   // causal mask
        if (kstart + 16 + lidx > row) v1 = -3.0e38f;
        float t = fmaxf(v0, v1);
        t = fmaxf(t, __shfl_xor(t, 1));
        t = fmaxf(t, __shfl_xor(t, 2));
        t = fmaxf(t, __shfl_xor(t, 4));
        t = fmaxf(t, __shfl_xor(t, 8));
        float mn    = fmaxf(mrow[r], t);
        float alpha = __expf(mrow[r] - mn);
        float p0 = __expf(v0 - mn), p1 = __expf(v1 - mn);
        float rs = p0 + p1;
        rs += __shfl_xor(rs, 1);
        rs += __shfl_xor(rs, 2);
        rs += __shfl_xor(rs, 4);
        rs += __shfl_xor(rs, 8);
        lrow[r] = lrow[r] * alpha + rs;
        mrow[r] = mn;
        p0a[r] = p0; p1a[r] = p1;
#pragma unroll
        for (int t2 = 0; t2 < 8; ++t2) o[t2][r] *= alpha;
      }

      // C-layout P -> LDS -> A-layout fragment (wave-internal transpose)
      _Float16* pw = &Ps[w * 16 * 32];
#pragma unroll
      for (int r = 0; r < 8; ++r) {
        int mrw = r + 8 * hi;
        pw[mrw * 32 + lidx]      = (_Float16)p0a[r];
        pw[mrw * 32 + 16 + lidx] = (_Float16)p1a[r];
      }
      asm volatile("s_wait_dscnt 0x0" ::: "memory");   // CDNA5 split DS counter
      __builtin_amdgcn_wave_barrier();

      union { v16h v; unsigned int u[8]; } ap;
#pragma unroll
      for (int j = 0; j < 8; ++j) {
        int kk = kb + ((j < 4) ? (j << 1) : (16 + ((j - 4) << 1)));
        ap.u[j] = *(const unsigned int*)(pw + lidx * 32 + kk);
      }

      // O += P * V  (8 N-tiles of 16 covering HD=128)
#pragma unroll
      for (int t2 = 0; t2 < 8; ++t2) {
        v16h bv;
#pragma unroll
        for (int s = 0; s < 16; ++s)
          bv[s] = Vs[(kb + s + ((s >> 3) << 3)) * 128 + t2 * 16 + lidx];
        o[t2] = __builtin_amdgcn_wmma_f32_16x16x32_f16(false, ap.v, false, bv,
                                                       (short)0, o[t2], false, false);
      }
    }
    __syncthreads();
  }

  // Normalize and store to [B, S, NH, HD] (== [B*S, NH*HD] for the O-proj GEMM)
#pragma unroll
  for (int t2 = 0; t2 < 8; ++t2)
#pragma unroll
    for (int r = 0; r < 8; ++r) {
      int srow = q0 + r + 8 * hi;
      int d    = t2 * 16 + lidx;
      float val = o[t2][r] / lrow[r];
      O[(((size_t)b * kS + srow) * kNH + h) * kHD + d] = (_Float16)val;
    }
}

// ---------------------------------------------------------------------------
// Launcher
// ---------------------------------------------------------------------------
extern "C" void kernel_launch(void* const* d_in, const int* in_sizes, int n_in,
                              void* d_out, int out_size, void* d_ws, size_t ws_size,
                              hipStream_t stream) {
  (void)in_sizes; (void)n_in; (void)out_size; (void)ws_size;
  const float* hs = (const float*)d_in[0];
  const int*  pos = (const int*)d_in[1];
  const float* Wq = (const float*)d_in[2];
  const float* bq = (const float*)d_in[3];
  const float* Wk = (const float*)d_in[4];
  const float* bk = (const float*)d_in[5];
  const float* Wv = (const float*)d_in[6];
  const float* bv = (const float*)d_in[7];
  const float* Wo = (const float*)d_in[8];
  float* out = (float*)d_out;

  char* ws = (char*)d_ws;
  size_t off = 0;
  auto alloc_h = [&](size_t elems) -> _Float16* {
    _Float16* p = (_Float16*)(ws + off);
    off = (off + elems * sizeof(_Float16) + 255) & ~(size_t)255;
    return p;
  };
  const size_t nXW = (size_t)kM * kH;           // 16.78M
  const size_t nKV = (size_t)kH * (kNKV * kHD); //  4.19M
  _Float16* Xh  = alloc_h(nXW);
  _Float16* Wqh = alloc_h(nXW);
  _Float16* Wkh = alloc_h(nKV);
  _Float16* Wvh = alloc_h(nKV);
  _Float16* Woh = alloc_h(nXW);
  _Float16* Qh  = alloc_h(nXW);                 // [B,NH,S,HD]
  _Float16* Kh  = alloc_h(nKV);                 // [B,NKV,S,HD]
  _Float16* Vh  = alloc_h(nKV);
  _Float16* Ah  = alloc_h(nXW);                 // attention out [B,S,NH,HD]

  auto cvt = [&](const float* s, _Float16* d, size_t n) {
    cvt_f32_f16_kernel<<<(unsigned)((n + 255) / 256), 256, 0, stream>>>(s, d, (int)n);
  };
  cvt(hs, Xh, nXW);
  cvt(Wq, Wqh, nXW);
  cvt(Wk, Wkh, nKV);
  cvt(Wv, Wvh, nKV);
  cvt(Wo, Woh, nXW);

  // QKV projections (WMMA + async LDS staging)
  gemm_wmma_kernel<0><<<dim3(kH / 128, kM / 128), 256, 0, stream>>>(
      Xh, Wqh, bq, Qh, nullptr, kM, kH, kH, kNH);
  gemm_wmma_kernel<0><<<dim3((kNKV * kHD) / 128, kM / 128), 256, 0, stream>>>(
      Xh, Wkh, bk, Kh, nullptr, kM, kNKV * kHD, kH, kNKV);
  gemm_wmma_kernel<0><<<dim3((kNKV * kHD) / 128, kM / 128), 256, 0, stream>>>(
      Xh, Wvh, bv, Vh, nullptr, kM, kNKV * kHD, kH, kNKV);

  // RoPE on Q and K
  int nq = kB * kNH * kS * (kHD / 2);
  int nk = kB * kNKV * kS * (kHD / 2);
  rope_kernel<<<(nq + 255) / 256, 256, 0, stream>>>(Qh, pos, kNH, nq);
  rope_kernel<<<(nk + 255) / 256, 256, 0, stream>>>(Kh, pos, kNKV, nk);

  // Flash attention (WMMA + TDM V staging)
  attn_wmma_kernel<<<kB * kNH * (kS / 64), 128, 0, stream>>>(Qh, Kh, Vh, Ah);

  // Output projection -> f32 result (WMMA)
  gemm_wmma_kernel<1><<<dim3(kH / 128, kM / 128), 256, 0, stream>>>(
      Ah, Woh, nullptr, nullptr, out, kM, kH, kH, 0);
}
